// PhysicsInformedModel_46514495816298
// MI455X (gfx1250) — matrix operand order repaired and checked by reading, not compiled, not run on hardware
//
#include <hip/hip_runtime.h>

// MI455X / gfx1250, wave32. Mixed precision: f16 WMMA operands, f32 accumulate,
// f32 tanh/derivative chain (branch-free HW transcendentals). 4 JVP streams
// (val, d/dz, d/dt, d2/dz2) share B (weight) fragments per WMMA tile.

typedef __attribute__((ext_vector_type(16))) _Float16 v16h;
typedef __attribute__((ext_vector_type(8)))  _Float16 v8h;
typedef __attribute__((ext_vector_type(8)))  float    v8f;

#define CAT16(lo, hi) __builtin_shufflevector((lo), (hi), \
    0, 1, 2, 3, 4, 5, 6, 7, 8, 9, 10, 11, 12, 13, 14, 15)

__device__ __forceinline__ v8f wmma_f32_f16(v16h a, v16h b, v8f c) {
  // D(16x16,f32) = A(16x32,f16) * B(32x16,f16) + C
  return __builtin_amdgcn_wmma_f32_16x16x32_f16(
      /*neg_a=*/false, a, /*neg_b=*/false, b,
      /*c_mod=*/(short)0, c, /*reuse_a=*/false, /*reuse_b=*/false);
}

// Branch-free tanh on the TRANS32 pipe. Native V_TANH_F32 if the toolchain
// exposes it; otherwise exp2+rcp (saturates correctly: exp2->inf/0 => +/-1).
__device__ __forceinline__ float fast_tanh(float x) {
#if __has_builtin(__builtin_amdgcn_tanhf)
  return __builtin_amdgcn_tanhf(x);
#else
  const float e = __builtin_amdgcn_exp2f(x * 2.8853900817779268f); // exp(2x)
  return 1.0f - 2.0f * __builtin_amdgcn_rcpf(e + 1.0f);
#endif
}

#define WAVES 4
#define PTS_PER_BLOCK (WAVES * 16)

__global__ __launch_bounds__(WAVES * 32, 1)
void pinn_mlp_jvp_kernel(const float* __restrict__ X,
                         const float* __restrict__ W1, const float* __restrict__ B1,
                         const float* __restrict__ W2, const float* __restrict__ B2,
                         const float* __restrict__ W3, const float* __restrict__ B3,
                         const float* __restrict__ W4, const float* __restrict__ B4,
                         float* __restrict__ OUT, int Ntot) {
  // ---- LDS ----
  __shared__ __align__(16) _Float16 wt2[128 * 128];           // W2^T : [out n][in k]
  __shared__ __align__(16) _Float16 wt3[64 * 128];            // W3^T : [out n][in k]
  __shared__ __align__(16) _Float16 acts2[WAVES][4][16][128]; // layer-2 out, [stream][pt][feat]
  __shared__ __align__(16) _Float16 hout3[WAVES][4][16][64];  // layer-3 out
  __shared__ float sW1[3 * 128];
  __shared__ float sb1[128], sb2[128], sb3[64], sW4[64];
  __shared__ float sb4;

  const int tid = threadIdx.x;

  // ---- one-time weight staging: transpose + f16 convert ----
  for (int i = tid; i < 128 * 128; i += blockDim.x) {
    int k = i & 127, n = i >> 7;                 // consecutive tid -> consecutive k (LDS-friendly)
    wt2[n * 128 + k] = (_Float16)W2[k * 128 + n];
  }
  for (int i = tid; i < 64 * 128; i += blockDim.x) {
    int k = i & 127, n = i >> 7;
    wt3[n * 128 + k] = (_Float16)W3[k * 64 + n];
  }
  for (int i = tid; i < 3 * 128; i += blockDim.x) sW1[i] = W1[i];
  for (int i = tid; i < 128; i += blockDim.x) { sb1[i] = B1[i]; sb2[i] = B2[i]; }
  for (int i = tid; i < 64;  i += blockDim.x) { sb3[i] = B3[i]; sW4[i] = W4[i]; }
  if (tid == 0) sb4 = B4[0];
  __syncthreads();

  const int lane   = tid & 31;
  const int wv     = tid >> 5;
  const int m      = lane & 15;      // point row within wave tile; also D/B column index
  const int hihalf = lane >> 4;      // 0: lanes 0-15, 1: lanes 16-31
  const int koff8  = hihalf * 8;     // A-fragment K offset per lane half
  const int koff16 = hihalf * 16;    // unused; B fragments use koff8 like A
  const int roff8  = hihalf * 8;     // D-fragment row offset per lane half

  int gpt = blockIdx.x * PTS_PER_BLOCK + wv * 16 + m;
  const int gptc = (gpt < Ntot) ? gpt : (Ntot - 1);   // clamp: keep EXEC all-ones for WMMA

  _Float16* __restrict__ myacts = &acts2[wv][0][0][0]; // stride: s*2048 + pt*128 + f
  _Float16* __restrict__ myout3 = &hout3[wv][0][0][0]; // stride: s*1024 + pt*64  + f

  // ================= layer 1 (3 -> 128), pure VALU, emit A-fragments =================
  // A-fragment (16-bit, 16x32): lane m holds K = 32q + 16*hh + koff8 + r in part p = 2q+hh, elem r.
  v8h Aval[8], Au[8], Av[8], Aw[8];
  {
    const float x0 = X[gptc * 3 + 0];
    const float x1 = X[gptc * 3 + 1];
    const float x2 = X[gptc * 3 + 2];
#pragma unroll
    for (int p = 0; p < 8; ++p) {
      const int q = p >> 1, hh = p & 1;
#pragma unroll
      for (int r = 0; r < 8; ++r) {
        const int f = 32 * q + 16 * hh + koff8 + r;
        const float w0 = sW1[f], w1 = sW1[128 + f], w2 = sW1[256 + f];
        const float a  = sb1[f] + x0 * w0 + x1 * w1 + x2 * w2;
        const float h  = fast_tanh(a);
        const float tp = 1.0f - h * h;            // sigma'
        // tangent pre-acts: a_u = W1[0][f], a_v = W1[1][f], a_w = 0
        Aval[p][r] = (_Float16)h;
        Au[p][r]   = (_Float16)(tp * w0);
        Av[p][r]   = (_Float16)(tp * w1);
        Aw[p][r]   = (_Float16)(-2.0f * h * tp * w0 * w0);  // sigma'' * a_u^2
      }
    }
  }

  // ================= layer 2 (128 -> 128), WMMA =================
  for (int ct = 0; ct < 8; ++ct) {                 // 16-col output tiles
    const float bb = sb2[ct * 16 + m];
    v8f dv = {bb, bb, bb, bb, bb, bb, bb, bb};     // C init with bias (col = lane%16)
    v8f du = {0, 0, 0, 0, 0, 0, 0, 0};
    v8f dt = {0, 0, 0, 0, 0, 0, 0, 0};
    v8f dw = {0, 0, 0, 0, 0, 0, 0, 0};
#pragma unroll
    for (int q = 0; q < 4; ++q) {                  // K chunks of 32
      const _Float16* wp = &wt2[(ct * 16 + m) * 128 + 32 * q + koff8];
      const v16h B = CAT16(*(const v8h*)wp, *(const v8h*)(wp + 16));
      dv = wmma_f32_f16(CAT16(Aval[2 * q], Aval[2 * q + 1]), B, dv);
      du = wmma_f32_f16(CAT16(Au[2 * q], Au[2 * q + 1]), B, du);
      dt = wmma_f32_f16(CAT16(Av[2 * q], Av[2 * q + 1]), B, dt);
      dw = wmma_f32_f16(CAT16(Aw[2 * q], Aw[2 * q + 1]), B, dw);
    }
    // tanh chain in f32, scatter D-layout -> [pt][feat] (per-wave LDS, in-order DS)
#pragma unroll
    for (int e = 0; e < 8; ++e) {
      const float h  = fast_tanh(dv[e]);
      const float tp = 1.0f - h * h;
      const float au = du[e];
      const int pr = e + roff8;                    // point row
      const int fc = ct * 16 + m;                  // feature column
      myacts[0 * 2048 + pr * 128 + fc] = (_Float16)h;
      myacts[1 * 2048 + pr * 128 + fc] = (_Float16)(tp * au);
      myacts[2 * 2048 + pr * 128 + fc] = (_Float16)(tp * dt[e]);
      myacts[3 * 2048 + pr * 128 + fc] = (_Float16)(tp * dw[e] - 2.0f * h * tp * au * au);
    }
  }

  // ================= layer 3 (128 -> 64), WMMA, A gathered from LDS =================
  for (int ct = 0; ct < 4; ++ct) {
    const float bb = sb3[ct * 16 + m];
    v8f dv = {bb, bb, bb, bb, bb, bb, bb, bb};
    v8f du = {0, 0, 0, 0, 0, 0, 0, 0};
    v8f dt = {0, 0, 0, 0, 0, 0, 0, 0};
    v8f dw = {0, 0, 0, 0, 0, 0, 0, 0};
#pragma unroll
    for (int q = 0; q < 4; ++q) {
      const _Float16* wp = &wt3[(ct * 16 + m) * 128 + 32 * q + koff8];
      const v16h B = CAT16(*(const v8h*)wp, *(const v8h*)(wp + 16));
#pragma unroll
      for (int s = 0; s < 4; ++s) {
        const _Float16* ap = &myacts[s * 2048 + m * 128 + 32 * q + koff8];
        const v16h A = CAT16(*(const v8h*)ap, *(const v8h*)(ap + 16));
        if (s == 0)      dv = wmma_f32_f16(A, B, dv);
        else if (s == 1) du = wmma_f32_f16(A, B, du);
        else if (s == 2) dt = wmma_f32_f16(A, B, dt);
        else             dw = wmma_f32_f16(A, B, dw);
      }
    }
#pragma unroll
    for (int e = 0; e < 8; ++e) {
      const float h  = fast_tanh(dv[e]);
      const float tp = 1.0f - h * h;
      const float au = du[e];
      const int pr = e + roff8;
      const int fc = ct * 16 + m;
      myout3[0 * 1024 + pr * 64 + fc] = (_Float16)h;
      myout3[1 * 1024 + pr * 64 + fc] = (_Float16)(tp * au);
      myout3[2 * 1024 + pr * 64 + fc] = (_Float16)(tp * dt[e]);
      myout3[3 * 1024 + pr * 64 + fc] = (_Float16)(tp * dw[e] - 2.0f * h * tp * au * au);
    }
  }

  // ================= layer 4 (64 -> 1): per-lane dot + cross-half reduce =================
  float res[4];
  const int fb = hihalf * 32;                      // lanes 0-15: feats 0-31; 16-31: feats 32-63
#pragma unroll
  for (int s = 0; s < 4; ++s) {
    const v8h* hp = (const v8h*)&myout3[s * 1024 + m * 64 + fb];
    float acc = 0.0f;
#pragma unroll
    for (int g = 0; g < 4; ++g) {                  // 4 x ds_load_b128
      const v8h hvec = hp[g];
#pragma unroll
      for (int j = 0; j < 8; ++j) acc += (float)hvec[j] * sW4[fb + g * 8 + j];
    }
    acc += __shfl_xor(acc, 16);                    // wave32: combine lane halves
    res[s] = acc;
  }
  res[0] += sb4;

  if (lane < 16 && gpt < Ntot) {
    float4 o;
    o.x = res[0];  // T
    o.y = res[1];  // dT/dz
    o.z = res[2];  // dT/dt
    o.w = res[3];  // d2T/dz2
    ((float4*)OUT)[gpt] = o;
  }
}

extern "C" void kernel_launch(void* const* d_in, const int* in_sizes, int n_in,
                              void* d_out, int out_size, void* d_ws, size_t ws_size,
                              hipStream_t stream) {
  const float* X  = (const float*)d_in[0];
  const float* W1 = (const float*)d_in[1];
  const float* B1 = (const float*)d_in[2];
  const float* W2 = (const float*)d_in[3];
  const float* B2 = (const float*)d_in[4];
  const float* W3 = (const float*)d_in[5];
  const float* B3 = (const float*)d_in[6];
  const float* W4 = (const float*)d_in[7];
  const float* B4 = (const float*)d_in[8];

  const int N = in_sizes[0] / 3;
  const int blocks = (N + PTS_PER_BLOCK - 1) / PTS_PER_BLOCK;
  pinn_mlp_jvp_kernel<<<blocks, WAVES * 32, 0, stream>>>(
      X, W1, B1, W2, B2, W3, B3, W4, B4, (float*)d_out, N);
}
